// ArbitraryRNN_40467181863072
// MI455X (gfx1250) — matrix operand order repaired and checked by reading, not compiled, not run on hardware
//
#include <hip/hip_runtime.h>
#include <stdint.h>

// ---------------------------------------------------------------------------
// Problem constants (reference: T,B,I,H = 2048,64,256,256)
// ---------------------------------------------------------------------------
#define Tn 2048
#define Bn 64
#define Hn 256

typedef __attribute__((ext_vector_type(16))) __bf16 v16bf;
typedef __attribute__((ext_vector_type(8)))  float  v8f;
typedef __attribute__((ext_vector_type(4)))  unsigned int u32x4;
typedef __attribute__((ext_vector_type(8)))  int    i32x8;
typedef __attribute__((ext_vector_type(4)))  int    i32x4;

// ---------------------------------------------------------------------------
// Activation: CDNA5 has a hardware V_TANH transcendental.
// ---------------------------------------------------------------------------
static __device__ inline float act_tanh(float x) {
#if __has_builtin(__builtin_amdgcn_tanhf)
  return __builtin_amdgcn_tanhf(x);   // v_tanh_f32
#else
  return tanhf(x);
#endif
}

// ---------------------------------------------------------------------------
// WMMA fragment loaders (layouts per CDNA5 ISA 7.12.2, wave32)
// A: 16x32 bf16. lane l -> row (l&15); K base = 8*(l>>4); elements
//    e=0..7 -> K=kb+e, e=8..15 -> K=kb+16+(e-8). Two contiguous 16B runs.
// ---------------------------------------------------------------------------
static __device__ inline v16bf frag_a_bf16(const __bf16* p, int ld, int m0,
                                           int k0, int lane) {
  const int m  = m0 + (lane & 15);
  const int kb = k0 + ((lane >> 4) << 3);
  const __bf16* q = p + m * ld + kb;
  v16bf a;
#pragma unroll
  for (int e = 0; e < 8; ++e) { a[e] = q[e]; a[e + 8] = q[e + 16]; }
  return a;
}

static __device__ inline v16bf frag_a_f32(const float* p, int ld, int m0,
                                          int k0, int lane) {
  const int m  = m0 + (lane & 15);
  const int kb = k0 + ((lane >> 4) << 3);
  const float* q = p + m * ld + kb;
  v16bf a;
#pragma unroll
  for (int e = 0; e < 8; ++e) {
    a[e]     = (__bf16)q[e];
    a[e + 8] = (__bf16)q[e + 16];
  }
  return a;
}

// B: 32x16 bf16, B[k][n] = W[n][k] (W row-major [N][K]).
// lane l -> col n0+(l&15); K base = 8*(l>>4); e=0..7 -> kb+e, e=8..15 -> kb+16+(e-8).
static __device__ inline v16bf frag_b(const __bf16* w, int ld, int n0, int k0,
                                      int lane) {
  const int n  = n0 + (lane & 15);
  const int kb = k0 + ((lane >> 4) << 3);
  const __bf16* q = w + n * ld + kb;
  v16bf b;
#pragma unroll
  for (int e = 0; e < 8; ++e) { b[e] = q[e]; b[e + 8] = q[e + 16]; }
  return b;
}

static __device__ inline v16bf load_a(const __bf16* p, int ld, int m0, int k0,
                                      int lane) {
  return frag_a_bf16(p, ld, m0, k0, lane);
}
static __device__ inline v16bf load_a(const float* p, int ld, int m0, int k0,
                                      int lane) {
  return frag_a_f32(p, ld, m0, k0, lane);
}

// ---------------------------------------------------------------------------
// Input-projection GEMM: out[M=T*B][256] = A[M][256] * W^T + b_ih   (fp32 out)
// Block = 256 threads = 8 waves; block tile = 64(M) x 256(N).
// Wave w: M-tile (w>>1), N half (w&1)*128 -> 8 accumulators, A reused 8x.
// ---------------------------------------------------------------------------
template <typename TIN>
__global__ __launch_bounds__(256) void gemm_xg(const TIN* __restrict__ A,
                                               const __bf16* __restrict__ W,
                                               const float* __restrict__ bias,
                                               float* __restrict__ out) {
  const int tid  = threadIdx.x;
  const int wid  = tid >> 5;
  const int lane = tid & 31;
  const int m0   = blockIdx.x * 64 + (wid >> 1) * 16;
  const int nb   = (wid & 1) * 128;

  v8f c[8];
#pragma unroll
  for (int j = 0; j < 8; ++j)
#pragma unroll
    for (int r = 0; r < 8; ++r) c[j][r] = 0.0f;

#pragma unroll
  for (int k0 = 0; k0 < Hn; k0 += 32) {
    v16bf a = load_a(A, Hn, m0, k0, lane);
#pragma unroll
    for (int j = 0; j < 8; ++j) {
      v16bf b = frag_b(W, Hn, nb + j * 16, k0, lane);
      c[j] = __builtin_amdgcn_wmma_f32_16x16x32_bf16(
          false, a, false, b, (short)0, c[j], false, false);
    }
  }

  const int mh = (lane >> 4) << 3;  // D row offset: r + 8*(lane>=16)
#pragma unroll
  for (int j = 0; j < 8; ++j) {
    const int   n  = nb + j * 16 + (lane & 15);
    const float bn = bias[n];
#pragma unroll
    for (int r = 0; r < 8; ++r) {
      const int row = m0 + r + mh;
      out[(size_t)row * Hn + n] = c[j][r] + bn;
    }
  }
}

// ---------------------------------------------------------------------------
// Recurrent scan: h_t = tanh(xg_t + h_{t-1} * W_hh^T + b_hh)
// One block per 16 batch rows (batch independence -> 4 blocks).
// 512 threads = 16 waves; wave w owns N-tile w. W_hh (128KB bf16) is staged
// into LDS via the Tensor Data Mover; h is double-buffered (2 x 8KB) so only
// one barrier per timestep is needed. Two accumulator chains + depth-2
// fragment pipeline break the ds_load -> wmma serialization.
// ---------------------------------------------------------------------------
__global__ __launch_bounds__(512) void rnn_scan(const float* __restrict__ xg,
                                                const __bf16* __restrict__ whh,
                                                const float* __restrict__ bhh,
                                                __bf16* __restrict__ hseq,
                                                float* __restrict__ hfin) {
  extern __shared__ char smem[];
  __bf16* w_lds = (__bf16*)smem;  // 256*256 bf16 = 128KB @ LDS offset 0
  __bf16* hb0   = (__bf16*)(smem + (size_t)Hn * Hn * 2);           // 8KB
  __bf16* hb1   = (__bf16*)(smem + (size_t)Hn * Hn * 2 + 16 * Hn * 2);

  const int tid  = threadIdx.x;
  const int wid  = tid >> 5;
  const int lane = tid & 31;
  const int b0   = blockIdx.x * 16;  // batch slice

  // ---- Stage W_hh into LDS ------------------------------------------------
#if __has_builtin(__builtin_amdgcn_tensor_load_to_lds)
  if (wid == 0) {
    // Tensor DMA descriptor (D#), per CDNA5 ISA.
    // 2D tensor, data_size=2B, tensor 256x256, tile 256x256, dst LDS offset 0
    // (dynamic-LDS base; no static LDS in this kernel).
    const unsigned long long ga = (unsigned long long)whh;
    u32x4 g0;
    g0[0] = 1u;                                  // count=1, user mode
    g0[1] = 0u;                                  // lds_addr = 0
    g0[2] = (unsigned int)(ga & 0xFFFFFFFFull);  // global_addr[31:0]
    g0[3] = (unsigned int)((ga >> 32) & 0x1FFFFFFull) | (2u << 30);  // type=2
    i32x8 g1;
    g1[0] = 0x00010000;          // workgroup_mask=0, data_size=1 (2 bytes)
    g1[1] = (int)(256u << 16);   // tensor_dim0[15:0]=256 @ bits[63:48]
    g1[2] = (int)(256u << 16);   // dim0 hi=0, tensor_dim1[15:0]=256 @[95:80]
    g1[3] = (int)(256u << 16);   // dim1 hi=0, tile_dim0=256 @ bits[127:112]
    g1[4] = 256;                 // tile_dim1=256, tile_dim2=0
    g1[5] = 256;                 // tensor_dim0_stride = 256 (low 32)
    g1[6] = 0;                   // stride0 hi=0, stride1[15:0]=0 (65536&0xFFFF)
    g1[7] = 1;                   // tensor_dim1_stride[47:16] = 65536>>16
    i32x4 z4; z4[0] = z4[1] = z4[2] = z4[3] = 0;
#if defined(__clang_major__) && __clang_major__ >= 23
    i32x8 z8;
#pragma unroll
    for (int i = 0; i < 8; ++i) z8[i] = 0;
    __builtin_amdgcn_tensor_load_to_lds(g0, g1, z4, z4, z8, 0);
#else
    __builtin_amdgcn_tensor_load_to_lds(g0, g1, z4, z4, 0);
#endif
    __builtin_amdgcn_s_wait_tensorcnt(0);
  }
#else
  {
    const uint4* s = (const uint4*)whh;
    uint4*       d = (uint4*)w_lds;
    for (int i = tid; i < (Hn * Hn * 2) / 16; i += 512) d[i] = s[i];
  }
#endif
  // zero both h buffers (2 x 512 x 16B = 16KB)
  {
    uint4 z; z.x = z.y = z.z = z.w = 0u;
    ((uint4*)hb0)[tid] = z;
    ((uint4*)hb1)[tid] = z;
  }
  __syncthreads();

  const int   n0 = wid * 16;
  const int   n  = n0 + (lane & 15);
  const int   mh = (lane >> 4) << 3;
  const float bn = bhh[n];

  const __bf16* hin  = hb0;
  __bf16*       hout = hb1;

  for (int t = 0; t < Tn; ++t) {
    v8f c0, c1;
#pragma unroll
    for (int r = 0; r < 8; ++r) { c0[r] = 0.0f; c1[r] = 0.0f; }

    // depth-2 software pipeline, two independent accumulator chains
    v16bf fa0 = frag_a_bf16(hin, Hn, 0, 0, lane);
    v16bf fb0 = frag_b(w_lds, Hn, n0, 0, lane);
    v16bf fa1 = frag_a_bf16(hin, Hn, 0, 32, lane);
    v16bf fb1 = frag_b(w_lds, Hn, n0, 32, lane);
#pragma unroll
    for (int k0 = 0; k0 < Hn; k0 += 64) {
      v16bf fa2, fb2, fa3, fb3;
      if (k0 + 64 < Hn) {
        fa2 = frag_a_bf16(hin, Hn, 0, k0 + 64, lane);
        fb2 = frag_b(w_lds, Hn, n0, k0 + 64, lane);
        fa3 = frag_a_bf16(hin, Hn, 0, k0 + 96, lane);
        fb3 = frag_b(w_lds, Hn, n0, k0 + 96, lane);
      }
      c0 = __builtin_amdgcn_wmma_f32_16x16x32_bf16(
          false, fa0, false, fb0, (short)0, c0, false, false);
      c1 = __builtin_amdgcn_wmma_f32_16x16x32_bf16(
          false, fa1, false, fb1, (short)0, c1, false, false);
      fa0 = fa2; fb0 = fb2; fa1 = fa3; fb1 = fb3;
    }

    const float* xr = xg + (size_t)(t * Bn + b0) * Hn + n;
    float vals[8];
#pragma unroll
    for (int r = 0; r < 8; ++r)
      vals[r] = act_tanh(c0[r] + c1[r] + xr[(size_t)(r + mh) * Hn] + bn);

    if (t + 1 < Tn)  // global_prefetch_b8 for next step's xg slice
      __builtin_prefetch(xg + (size_t)((t + 1) * Bn + b0 + mh) * Hn + n, 0, 0);

#pragma unroll
    for (int r = 0; r < 8; ++r) {
      const int    bm = r + mh;
      const __bf16 hv = (__bf16)vals[r];
      hout[bm * Hn + n] = hv;
      if (hseq) hseq[(size_t)(t * Bn + b0 + bm) * Hn + n] = hv;
    }
    if (hfin && t == Tn - 1) {
#pragma unroll
      for (int r = 0; r < 8; ++r)
        hfin[(b0 + r + mh) * Hn + n] = vals[r];
    }
    __syncthreads();  // publish hout before next step reads it
    const __bf16* tmp = hin; hin = hout; hout = (__bf16*)tmp;
  }
}

// ---------------------------------------------------------------------------
// Weight prep: dst = bf16(src * mask) (mask==nullptr -> unmasked). 256x256.
// ---------------------------------------------------------------------------
__global__ __launch_bounds__(256) void prep_w(const float* __restrict__ src,
                                              const uint8_t* __restrict__ mask,
                                              __bf16* __restrict__ dst) {
  const int i = blockIdx.x * 256 + threadIdx.x;
  float v = src[i];
  if (mask) v = mask[i] ? v : 0.0f;
  dst[i] = (__bf16)v;
}

__global__ __launch_bounds__(256) void final_add(const float* __restrict__ a,
                                                 const float* __restrict__ b,
                                                 float* __restrict__ out) {
  const int i = blockIdx.x * 256 + threadIdx.x;
  out[i] = a[i] + b[i];
}

// ---------------------------------------------------------------------------
// Orchestration.
// Inputs: 0:x, then per layer l in {0,1,2,s}: w_ih,w_hh,b_ih,b_hh (1..16),
// then 17:mask1, 18:mask2, 19:mask_skip (jax bool -> 1 byte each).
// ---------------------------------------------------------------------------
extern "C" void kernel_launch(void* const* d_in, const int* in_sizes, int n_in,
                              void* d_out, int out_size, void* d_ws,
                              size_t ws_size, hipStream_t stream) {
  (void)in_sizes; (void)n_in; (void)out_size; (void)ws_size;

  const float* x = (const float*)d_in[0];
  const float* wih[4] = {(const float*)d_in[1], (const float*)d_in[5],
                         (const float*)d_in[9], (const float*)d_in[13]};
  const float* whh[4] = {(const float*)d_in[2], (const float*)d_in[6],
                         (const float*)d_in[10], (const float*)d_in[14]};
  const float* bih[4] = {(const float*)d_in[3], (const float*)d_in[7],
                         (const float*)d_in[11], (const float*)d_in[15]};
  const float* bhh[4] = {(const float*)d_in[4], (const float*)d_in[8],
                         (const float*)d_in[12], (const float*)d_in[16]};
  const uint8_t* msk[4] = {nullptr, (const uint8_t*)d_in[17],
                           (const uint8_t*)d_in[18], (const uint8_t*)d_in[19]};

  // Workspace carve-up (~257 MB)
  char*  ws  = (char*)d_ws;
  size_t off = 0;
  __bf16* h0s = (__bf16*)(ws + off); off += (size_t)Tn * Bn * Hn * 2;  // 64MB
  __bf16* h1s = (__bf16*)(ws + off); off += (size_t)Tn * Bn * Hn * 2;  // 64MB
  float*  xg  = (float*)(ws + off);  off += (size_t)Tn * Bn * Hn * 4;  // 128MB
  __bf16* Wih_bf[4]; __bf16* Whh_bf[4];
  for (int l = 0; l < 4; ++l) {
    Wih_bf[l] = (__bf16*)(ws + off); off += (size_t)Hn * Hn * 2;
    Whh_bf[l] = (__bf16*)(ws + off); off += (size_t)Hn * Hn * 2;
  }
  float* h2f = (float*)(ws + off); off += (size_t)Bn * Hn * 4;
  float* hsf = (float*)(ws + off); off += (size_t)Bn * Hn * 4;

  // 1) Mask + convert weights to bf16 (tiny)
  for (int l = 0; l < 4; ++l) {
    prep_w<<<(Hn * Hn) / 256, 256, 0, stream>>>(wih[l], msk[l], Wih_bf[l]);
    prep_w<<<(Hn * Hn) / 256, 256, 0, stream>>>(whh[l], nullptr, Whh_bf[l]);
  }

  const dim3   gemmGrid((Tn * Bn) / 64);
  const size_t scanLds =
      (size_t)Hn * Hn * 2 + 2 * (size_t)16 * Hn * 2;  // 128KB + 16KB

  // 2) Layer 0: xg0 = x * W_ih0^T + b_ih0 ; scan -> h0s (bf16 seq)
  gemm_xg<float><<<gemmGrid, 256, 0, stream>>>(x, Wih_bf[0], bih[0], xg);
  rnn_scan<<<4, 512, scanLds, stream>>>(xg, Whh_bf[0], bhh[0], h0s, nullptr);

  // 3) Layer 1 (mask1): scan -> h1s
  gemm_xg<__bf16><<<gemmGrid, 256, 0, stream>>>(h0s, Wih_bf[1], bih[1], xg);
  rnn_scan<<<4, 512, scanLds, stream>>>(xg, Whh_bf[1], bhh[1], h1s, nullptr);

  // 4) Layer 2 (mask2): only final timestep needed -> h2f
  gemm_xg<__bf16><<<gemmGrid, 256, 0, stream>>>(h1s, Wih_bf[2], bih[2], xg);
  rnn_scan<<<4, 512, scanLds, stream>>>(xg, Whh_bf[2], bhh[2], nullptr, h2f);

  // 5) Skip RNN (mask_skip) from h0s: only final timestep -> hsf
  gemm_xg<__bf16><<<gemmGrid, 256, 0, stream>>>(h0s, Wih_bf[3], bih[3], xg);
  rnn_scan<<<4, 512, scanLds, stream>>>(xg, Whh_bf[3], bhh[3], nullptr, hsf);

  // 6) out = h2f + hsf  ([B,H] fp32)
  final_add<<<(Bn * Hn) / 256, 256, 0, stream>>>(h2f, hsf, (float*)d_out);
}
